// LocalConvLayer_8650064134372
// MI455X (gfx1250) — matrix operand order repaired and checked by reading, not compiled, not run on hardware
//
#include <hip/hip_runtime.h>
#include <hip/hip_bf16.h>

typedef __attribute__((ext_vector_type(16))) _Float16 v16h;
typedef __attribute__((ext_vector_type(8)))  _Float16 v8h;
typedef __attribute__((ext_vector_type(8)))  float    v8f;

#define WMMA_F16(A, B, C) \
  __builtin_amdgcn_wmma_f32_16x16x32_f16(false, (A), false, (B), (short)0, (C), false, false)

constexpr int B_ = 2, L_ = 1024, D_ = 256, H_ = 64, C_ = 4;

// ---------------------------------------------------------------------------
// Kernel 0: embed (f32, masked) -> f16
// ---------------------------------------------------------------------------
__global__ __launch_bounds__(256) void prep_kernel(
    const float* __restrict__ embed, const unsigned char* __restrict__ mask,
    _Float16* __restrict__ Eh, int total) {
  int e = blockIdx.x * 256 + threadIdx.x;
  if (e < total) {
    int rl = e / D_;  // flattened (b,l)
    Eh[e] = mask[rl] ? (_Float16)0.0f : (_Float16)embed[e];
  }
}

// ---------------------------------------------------------------------------
// Kernel 1: SIREN kernel matrix.  Each wave = 16 taus (fixed i, 16 j's).
// Hidden layers: 16x64x64 GEMMs (8 WMMAs each); head 64->4 is a WMMA against
// Wk^T zero-padded to 64x16.  Output: Kbuf (B,C,L,L) f16, zero outside the
// causal/horizon window (reference sets tau=-1 there -> masked).
// ---------------------------------------------------------------------------
__global__ __launch_bounds__(256) void siren_kernel(
    const float* __restrict__ time_seq, const unsigned char* __restrict__ mask,
    const float* __restrict__ W0, const float* __restrict__ b0,
    const float* __restrict__ W1, const float* __restrict__ b1,
    const float* __restrict__ W2, const float* __restrict__ b2,
    const float* __restrict__ Wk, const float* __restrict__ bk,
    _Float16* __restrict__ Kbuf) {
  __shared__ _Float16 sW1[H_ * H_];
  __shared__ _Float16 sW2[H_ * H_];
  __shared__ _Float16 sWkT[16 * H_];  // [n][k], n>=4 zero-padded
  __shared__ float sW0[H_], sb0[H_], sb1[H_], sb2[H_], sbk[C_];
  __shared__ _Float16 h1buf[8][16 * H_];  // per-wave activation tiles
  __shared__ _Float16 h2buf[8][16 * H_];
  __shared__ float taubuf[8][16];

  const int tid  = threadIdx.x;
  const int wave = tid >> 5;
  const int lane = tid & 31;
  const int m    = lane & 15;   // tau row within the 16-tau group
  const int half = lane >> 4;   // lane-half selects K interleave (ISA 7.12.2)

  // block decode: 8 j-tiles of 128, 1024 i's, 2 batches
  const int bid = blockIdx.x;
  const int jt  = bid & 7;
  const int i   = (bid >> 3) & (L_ - 1);
  const int b   = bid >> 13;
  const int j0  = jt * 128 + wave * 16;

  // stage weights to LDS (f16 for WMMA B-operands)
  for (int e = tid; e < H_ * H_; e += 256) {
    sW1[e] = (_Float16)W1[e];
    sW2[e] = (_Float16)W2[e];
  }
  for (int e = tid; e < 16 * H_; e += 256) {
    const int n = e >> 6, k = e & 63;
    sWkT[e] = (n < C_) ? (_Float16)Wk[n * H_ + k] : (_Float16)0.0f;
  }
  if (tid < H_) { sW0[tid] = W0[tid]; sb0[tid] = b0[tid]; sb1[tid] = b1[tid]; sb2[tid] = b2[tid]; }
  if (tid < C_) sbk[tid] = bk[tid];
  __syncthreads();

  // rel_tau for this lane's row (reference: future/masked -> -1)
  const int j   = j0 + m;
  const float ti = time_seq[b * L_ + i];
  const float tj = time_seq[b * L_ + j];
  const bool bad = (j > i) || mask[b * L_ + i] || mask[b * L_ + j];
  const float tau = bad ? -1.0f : (ti - tj);
  if (lane < 16) taubuf[wave][m] = tau;  // stash per-row tau for the head

  // layer 0: sin(W0*tau + b0) built directly in A-fragment layout.
  // lane<16 holds row m with K {0..15}; lane>=16 same row, K {16..31}.
  v16h a0, a1;
#pragma unroll
  for (int s = 0; s < 16; ++s) {
    const int kl = half * 16 + s;
    const int k0 = kl, k1 = 32 + kl;
    a0[s] = (_Float16)__sinf(sW0[k0] * tau + sb0[k0]);
    a1[s] = (_Float16)__sinf(sW0[k1] * tau + sb0[k1]);
  }

  _Float16* h1 = h1buf[wave];
  _Float16* h2 = h2buf[wave];

  // ---- layer 1: 16x64 @ 64x64  (8 WMMAs) ----
  v8f c1[4];
#pragma unroll
  for (int nb = 0; nb < 4; ++nb) {
    const int n = nb * 16 + m;  // B-frag: col n across lanes, 16 contiguous K per half
    const v16h bk0 = *(const v16h*)&sW1[n * H_ + half * 16];
    const v16h bk1 = *(const v16h*)&sW1[n * H_ + 32 + half * 16];
    v8f c = {};
    c = WMMA_F16(a0, bk0, c);
    c = WMMA_F16(a1, bk1, c);
    c1[nb] = c;
  }
#pragma unroll
  for (int nb = 0; nb < 4; ++nb) {
    const int n = nb * 16 + m;
    const float bias = sb1[n];
#pragma unroll
    for (int r = 0; r < 8; ++r)  // C-layout: row = r + 8*half
      h1[(r + half * 8) * H_ + n] = (_Float16)__sinf(c1[nb][r] + bias);
  }

  // reload h1 as A fragments (same-wave LDS ops are in-order: no barrier)
  v16h ha0, ha1;
  ((v8h*)&ha0)[0] = *(const v8h*)&h1[m * H_ + half * 16];
  ((v8h*)&ha0)[1] = *(const v8h*)&h1[m * H_ + half * 16 + 8];
  ((v8h*)&ha1)[0] = *(const v8h*)&h1[m * H_ + 32 + half * 16];
  ((v8h*)&ha1)[1] = *(const v8h*)&h1[m * H_ + 32 + half * 16 + 8];

  // ---- layer 2 ----
  v8f c2[4];
#pragma unroll
  for (int nb = 0; nb < 4; ++nb) {
    const int n = nb * 16 + m;
    const v16h bk0 = *(const v16h*)&sW2[n * H_ + half * 16];
    const v16h bk1 = *(const v16h*)&sW2[n * H_ + 32 + half * 16];
    v8f c = {};
    c = WMMA_F16(ha0, bk0, c);
    c = WMMA_F16(ha1, bk1, c);
    c2[nb] = c;
  }
#pragma unroll
  for (int nb = 0; nb < 4; ++nb) {
    const int n = nb * 16 + m;
    const float bias = sb2[n];
#pragma unroll
    for (int r = 0; r < 8; ++r)
      h2[(r + half * 8) * H_ + n] = (_Float16)__sinf(c2[nb][r] + bias);
  }

  // ---- head: 16x64 @ 64x16 (Wk^T zero-padded) -> 2 WMMAs ----
  v16h hb0, hb1;
  ((v8h*)&hb0)[0] = *(const v8h*)&h2[m * H_ + half * 16];
  ((v8h*)&hb0)[1] = *(const v8h*)&h2[m * H_ + half * 16 + 8];
  ((v8h*)&hb1)[0] = *(const v8h*)&h2[m * H_ + 32 + half * 16];
  ((v8h*)&hb1)[1] = *(const v8h*)&h2[m * H_ + 32 + half * 16 + 8];
  const v16h bw0 = *(const v16h*)&sWkT[m * H_ + half * 16];
  const v16h bw1 = *(const v16h*)&sWkT[m * H_ + 32 + half * 16];
  v8f ck = {};
  ck = WMMA_F16(hb0, bw0, ck);
  ck = WMMA_F16(hb1, bw1, ck);

  // lanes with N = (lane&15) < 4 hold channel N for rows r + 8*half.
  if (m < C_) {
    const int ch = m;
    const float hor = (float)(1 << ch);  // HORIZON = {1,2,4,8}
    const float bias = sbk[ch];
    const size_t obase = ((size_t)(b * C_ + ch) * L_ + i) * L_ + j0;
#pragma unroll
    for (int r = 0; r < 8; ++r) {
      const int row = r + half * 8;
      const float taur = taubuf[wave][row];
      const float kv = (taur < 0.0f || taur > hor) ? 0.0f : (ck[r] + bias);
      Kbuf[obase + row] = (_Float16)kv;
    }
  }
}

// ---------------------------------------------------------------------------
// Kernel 2: X[b,:,c,:] = K[b,c] @ E[b]   (causal-trimmed f16 WMMA GEMM)
// block tile 64x128, 8 waves (4x2), 4 accumulators/wave, K-chunks of 32.
// ---------------------------------------------------------------------------
__global__ __launch_bounds__(256) void kconv_kernel(
    const _Float16* __restrict__ Kbuf,  // (B,C,L,L)
    const _Float16* __restrict__ Eh,    // (B,L,D)
    _Float16* __restrict__ Xh) {        // (B,L,C*D)
  __shared__ _Float16 sA[64 * 32];
  __shared__ _Float16 sB[128 * 32];  // E tile transposed: [d][j]

  const int bid = blockIdx.x;
  const int dt = bid & 1;
  const int it = (bid >> 1) & 15;
  const int c  = (bid >> 5) & 3;
  const int b  = bid >> 7;
  const int i0 = it * 64, d0 = dt * 128;

  const int tid = threadIdx.x, wave = tid >> 5, lane = tid & 31;
  const int mi = wave & 3, ni = wave >> 2;
  const int m = lane & 15, half = lane >> 4;

  v8f acc[4] = {};
  const size_t kbase = ((size_t)(b * C_ + c) * L_ + i0) * L_;
  const size_t ebase = (size_t)b * L_ * D_;

  const int nchunks = i0 / 32 + 2;  // rows beyond i are exact zeros -> skip
  for (int ch = 0; ch < nchunks; ++ch) {
    const int jb = ch * 32;
    __syncthreads();
    {  // stage A: 64 rows x 32 cols of K (contiguous 16B chunks)
      const int row = tid >> 2, seg = tid & 3;
      *(v8h*)&sA[row * 32 + seg * 8] =
          *(const v8h*)&Kbuf[kbase + (size_t)row * L_ + jb + seg * 8];
      if (ch + 1 < nchunks)
        __builtin_prefetch(&Kbuf[kbase + (size_t)row * L_ + jb + 32 + seg * 8], 0, 0);
    }
#pragma unroll
    for (int r = 0; r < 16; ++r) {  // stage E transposed
      const int d = tid & 127;
      const int jj = r * 2 + (tid >> 7);
      sB[d * 32 + jj] = Eh[ebase + (size_t)(jb + jj) * D_ + d0 + d];
    }
    __syncthreads();

    v16h a;
    ((v8h*)&a)[0] = *(const v8h*)&sA[(mi * 16 + m) * 32 + half * 16];
    ((v8h*)&a)[1] = *(const v8h*)&sA[(mi * 16 + m) * 32 + half * 16 + 8];
#pragma unroll
    for (int nb = 0; nb < 4; ++nb) {
      const v16h bf = *(const v16h*)&sB[(ni * 64 + nb * 16 + m) * 32 + half * 16];
      acc[nb] = WMMA_F16(a, bf, acc[nb]);
    }
  }

#pragma unroll
  for (int nb = 0; nb < 4; ++nb)
#pragma unroll
    for (int r = 0; r < 8; ++r) {
      const int irow = i0 + mi * 16 + r + half * 8;
      const int d = d0 + ni * 64 + nb * 16 + m;
      Xh[((size_t)b * L_ + irow) * (C_ * D_) + c * D_ + d] = (_Float16)acc[nb][r];
    }
}

// ---------------------------------------------------------------------------
// Kernel 3: Y = Xh @ Wout^T + bout + embed   (M=2048, K=1024, N=256)
// ---------------------------------------------------------------------------
__global__ __launch_bounds__(256) void proj_kernel(
    const _Float16* __restrict__ Xh,   // (B*L, C*D)
    const float* __restrict__ Wout,    // (D, C*D)
    const float* __restrict__ bout,
    const float* __restrict__ embed,   // (B,L,D) f32
    const unsigned char* __restrict__ mask,
    float* __restrict__ Y) {           // (B*L, D)
  __shared__ _Float16 sA[64 * 32];
  __shared__ _Float16 sW[128 * 32];  // Wt[n][k] : B[k][n]=Wout[n][k], contiguous in k

  const int bid = blockIdx.x;  // 32 m-tiles x 2 n-tiles
  const int nt = bid & 1;
  const int mt = bid >> 1;
  const int m0 = mt * 64, n0 = nt * 128;

  const int tid = threadIdx.x, wave = tid >> 5, lane = tid & 31;
  const int mi = wave & 3, ni = wave >> 2;
  const int m = lane & 15, half = lane >> 4;

  v8f acc[4] = {};
  for (int ch = 0; ch < (C_ * D_) / 32; ++ch) {
    const int kb = ch * 32;
    __syncthreads();
    {
      const int row = tid >> 2, seg = tid & 3;
      *(v8h*)&sA[row * 32 + seg * 8] =
          *(const v8h*)&Xh[((size_t)m0 + row) * (C_ * D_) + kb + seg * 8];
    }
#pragma unroll
    for (int r = 0; r < 16; ++r) {
      const int k = tid & 31;
      const int n = r * 8 + (tid >> 5);
      sW[n * 32 + k] = (_Float16)Wout[(size_t)(n0 + n) * (C_ * D_) + kb + k];
      if (ch + 1 < (C_ * D_) / 32)
        __builtin_prefetch(&Wout[(size_t)(n0 + n) * (C_ * D_) + kb + 32 + k], 0, 0);
    }
    __syncthreads();

    v16h a;
    ((v8h*)&a)[0] = *(const v8h*)&sA[(mi * 16 + m) * 32 + half * 16];
    ((v8h*)&a)[1] = *(const v8h*)&sA[(mi * 16 + m) * 32 + half * 16 + 8];
#pragma unroll
    for (int nb = 0; nb < 4; ++nb) {
      const v16h bf = *(const v16h*)&sW[(ni * 64 + nb * 16 + m) * 32 + half * 16];
      acc[nb] = WMMA_F16(a, bf, acc[nb]);
    }
  }

#pragma unroll
  for (int nb = 0; nb < 4; ++nb)
#pragma unroll
    for (int r = 0; r < 8; ++r) {
      const int row = m0 + mi * 16 + r + half * 8;  // flattened (b,l)
      const int d = n0 + ni * 64 + nb * 16 + m;
      const float e = mask[row] ? 0.0f : embed[(size_t)row * D_ + d];
      Y[(size_t)row * D_ + d] = acc[nb][r] + bout[d] + e;
    }
}

// ---------------------------------------------------------------------------
// Kernel 4: per-row LayerNorm
// ---------------------------------------------------------------------------
__global__ __launch_bounds__(256) void ln_kernel(
    const float* __restrict__ Y, const float* __restrict__ gamma,
    const float* __restrict__ beta, float* __restrict__ out) {
  __shared__ float rs[256], rs2[256];
  const int row = blockIdx.x, d = threadIdx.x;
  const float y = Y[(size_t)row * D_ + d];
  rs[d] = y;
  rs2[d] = y * y;
  __syncthreads();
  for (int s = 128; s > 0; s >>= 1) {
    if (d < s) { rs[d] += rs[d + s]; rs2[d] += rs2[d + s]; }
    __syncthreads();
  }
  const float mu = rs[0] * (1.0f / D_);
  const float var = rs2[0] * (1.0f / D_) - mu * mu;
  out[(size_t)row * D_ + d] = (y - mu) * rsqrtf(var + 1e-5f) * gamma[d] + beta[d];
}

// ---------------------------------------------------------------------------
extern "C" void kernel_launch(void* const* d_in, const int* in_sizes, int n_in,
                              void* d_out, int out_size, void* d_ws, size_t ws_size,
                              hipStream_t stream) {
  (void)in_sizes; (void)n_in; (void)out_size; (void)ws_size;
  const float* embed    = (const float*)d_in[0];
  const float* time_seq = (const float*)d_in[1];
  const unsigned char* mask = (const unsigned char*)d_in[2];
  const float* W0 = (const float*)d_in[3];  const float* b0v = (const float*)d_in[4];
  const float* W1 = (const float*)d_in[5];  const float* b1v = (const float*)d_in[6];
  const float* W2 = (const float*)d_in[7];  const float* b2v = (const float*)d_in[8];
  const float* Wk = (const float*)d_in[9];  const float* bkv = (const float*)d_in[10];
  const float* Wout = (const float*)d_in[11]; const float* bout = (const float*)d_in[12];
  const float* gamma = (const float*)d_in[13]; const float* beta = (const float*)d_in[14];
  float* out = (float*)d_out;

  char* ws = (char*)d_ws;
  _Float16* Eh   = (_Float16*)(ws);                                     // 1 MB
  _Float16* Kbuf = (_Float16*)(ws + (1 << 20));                         // 16 MB
  _Float16* Xh   = (_Float16*)(ws + (1 << 20) + (16 << 20));            // 4 MB
  float*    Y    = (float*)(ws + (1 << 20) + (16 << 20) + (4 << 20));   // 2 MB

  prep_kernel<<<(B_ * L_ * D_ + 255) / 256, 256, 0, stream>>>(embed, mask, Eh, B_ * L_ * D_);
  siren_kernel<<<B_ * L_ * (L_ / 128), 256, 0, stream>>>(
      time_seq, mask, W0, b0v, W1, b1v, W2, b2v, Wk, bkv, Kbuf);
  kconv_kernel<<<B_ * C_ * (L_ / 64) * (D_ / 128), 256, 0, stream>>>(Kbuf, Eh, Xh);
  proj_kernel<<<(B_ * L_ / 64) * (D_ / 128), 256, 0, stream>>>(Xh, Wout, bout, embed, mask, Y);
  ln_kernel<<<B_ * L_, 256, 0, stream>>>(Y, gamma, beta, out);
}
